// MultiHeadAttention_88622355186026
// MI455X (gfx1250) — matrix operand order repaired and checked by reading, not compiled, not run on hardware
//
#include <hip/hip_runtime.h>

typedef __attribute__((ext_vector_type(16))) _Float16 v16h;
typedef __attribute__((ext_vector_type(8)))  _Float16 v8h;
typedef __attribute__((ext_vector_type(8)))  float    v8f;

typedef __attribute__((ext_vector_type(4))) unsigned int u32x4;
typedef __attribute__((ext_vector_type(8))) int          i32x8;
typedef __attribute__((ext_vector_type(4))) int          i32x4;

#define WMMA_F32_F16(a, b, c) \
  __builtin_amdgcn_wmma_f32_16x16x32_f16(false, (a), false, (b), (short)0, (c), false, false)

#if __has_builtin(__builtin_amdgcn_tensor_load_to_lds) && \
    __has_builtin(__builtin_amdgcn_s_wait_tensorcnt)
#define HAVE_TDM 1
#else
#define HAVE_TDM 0
#endif

// Raw LDS byte offset of a __shared__ object (generic -> addrspace(3) ptrtoint).
__device__ __forceinline__ unsigned lds_addr_of(const void* p) {
  return (unsigned)(unsigned long long)(const __attribute__((address_space(3))) void*)p;
}

#if HAVE_TDM
// 2D tile DMA: copy tile1 rows x tile0 contiguous f16 elements (row stride
// stride0 elements) from global to LDS at lds_off. D# per CDNA5 ISA 8.3/8.4.
__device__ __forceinline__ void tdm_load_2d_f16(unsigned lds_off, const void* gaddr,
                                                unsigned tile0, unsigned tile1,
                                                unsigned stride0) {
  unsigned long long ga = (unsigned long long)gaddr;
  u32x4 g0;
  g0.x = 1u;                                   // count=1, user descriptor
  g0.y = lds_off;                              // lds_addr (bytes)
  g0.z = (unsigned)(ga & 0xFFFFFFFFu);         // global_addr[31:0]
  g0.w = (unsigned)((ga >> 32) & 0x01FFFFFFu)  // global_addr[56:32]
         | 0x80000000u;                        // type=2 ("image")
  i32x8 g1;
  g1[0] = (int)0x00010000u;                                  // data_size=2B
  g1[1] = (int)((tile0 & 0xFFFFu) << 16);                    // tensor_dim0 = tile0
  g1[2] = (int)((tile0 >> 16) | ((tile1 & 0xFFFFu) << 16));  // tensor_dim1 = tile1
  g1[3] = (int)((tile1 >> 16) | ((tile0 & 0xFFFFu) << 16));  // tile_dim0
  g1[4] = (int)(tile1 & 0xFFFFu);                            // tile_dim1 (tile_dim2=0)
  g1[5] = (int)stride0;                                      // tensor_dim0_stride[31:0]
  g1[6] = 0;                                                 // stride[47:32] | d1stride lo
  g1[7] = 0;
  i32x4 z4 = {};
#if defined(__clang_major__) && (__clang_major__ >= 23)
  i32x8 z8 = {};
  __builtin_amdgcn_tensor_load_to_lds(g0, g1, z4, z4, z8, 0);
#else
  __builtin_amdgcn_tensor_load_to_lds(g0, g1, z4, z4, 0);
#endif
}
#endif

// Build a 16x32 f16 A-fragment for one lane from a 32-wide row of LDS.
// Lane (hi,ln) element e holds K = hi*16 + e, row M = ln.
__device__ __forceinline__ v16h make_a_frag(const _Float16* row32, int hi) {
  v8h lo = *(const v8h*)(row32 + hi * 16);
  v8h hh = *(const v8h*)(row32 + hi * 16 + 8);
  return __builtin_shufflevector(lo, hh, 0, 1, 2, 3, 4, 5, 6, 7,
                                 8, 9, 10, 11, 12, 13, 14, 15);
}

// ---------------------------------------------------------------------------
// Generic WMMA GEMM: C[M,N] = A[M,K] * B[K,N] (+ bias), A/B row-major.
// Block = 256 threads (8 waves). Block tile 128x128, wave tile 32x64, KB=32.
// If TA == f16, the A tile is staged by the Tensor Data Mover.
// If SPLIT_V, columns >= 1536 (the V third of QKV) are scattered into Cv with
// layout [b][h][d][s] so attention can TDM V tiles pre-transposed.
// ---------------------------------------------------------------------------
template <typename TA, typename TB, typename TC, bool HAS_BIAS, bool SPLIT_V>
__global__ __launch_bounds__(256) void gemm_wmma_kernel(
    const TA* __restrict__ A, const TB* __restrict__ B, TC* __restrict__ C,
    TC* __restrict__ Cv, const float* __restrict__ bias, int M, int N, int K,
    int lda, int ldb, int ldc) {
  (void)M; (void)N;
  constexpr bool A_IS_F16 = __is_same(TA, _Float16);
  __shared__ __align__(32) _Float16 As[128][32];
  __shared__ __align__(32) _Float16 Bt[128][32];  // Bt[n][k]

  const int t    = threadIdx.x;
  const int lane = t & 31;
  const int wid  = t >> 5;
  const int hi   = lane >> 4;
  const int ln   = lane & 15;
  const int m0   = blockIdx.y * 128;
  const int n0   = blockIdx.x * 128;
  const int wm   = (wid >> 1) * 32;
  const int wn   = (wid & 1) * 64;

  v8f acc[2][4] = {};

  for (int kb = 0; kb < K; kb += 32) {
    __syncthreads();
#if HAVE_TDM
    if constexpr (A_IS_F16) {
      if (wid == 0)
        tdm_load_2d_f16(lds_addr_of(&As[0][0]),
                        A + (size_t)m0 * lda + kb, 32, 128, (unsigned)lda);
    }
#endif
    if (!(HAVE_TDM && A_IS_F16)) {
      // Stage A tile 128x32 (coalesced along k), convert to f16.
#pragma unroll
      for (int i = 0; i < 16; ++i) {
        int linear = i * 256 + t;
        int r = linear >> 5, c = linear & 31;
        As[r][c] = (_Float16)A[(size_t)(m0 + r) * lda + kb + c];
      }
    }
    // Stage B tile 32x128 transposed (coalesced along n).
#pragma unroll
    for (int i = 0; i < 16; ++i) {
      int linear = i * 256 + t;
      int kr = linear >> 7, c = linear & 127;
      Bt[c][kr] = (_Float16)B[(size_t)(kb + kr) * ldb + n0 + c];
    }
#if HAVE_TDM
    if constexpr (A_IS_F16) {
      if (wid == 0) __builtin_amdgcn_s_wait_tensorcnt(0);
    }
#endif
    __syncthreads();

    v16h af[2], bf[4];
#pragma unroll
    for (int mi = 0; mi < 2; ++mi)
      af[mi] = make_a_frag(&As[wm + mi * 16 + ln][0], hi);
#pragma unroll
    for (int ni = 0; ni < 4; ++ni)
      bf[ni] = *(const v16h*)&Bt[wn + ni * 16 + ln][hi * 16];
#pragma unroll
    for (int mi = 0; mi < 2; ++mi)
#pragma unroll
      for (int ni = 0; ni < 4; ++ni)
        acc[mi][ni] = WMMA_F32_F16(af[mi], bf[ni], acc[mi][ni]);
  }

  // Epilogue: C layout is row m = hi*8 + r, col n = ln per 16x16 tile.
#pragma unroll
  for (int mi = 0; mi < 2; ++mi) {
#pragma unroll
    for (int ni = 0; ni < 4; ++ni) {
      int col = n0 + wn + ni * 16 + ln;
      float badd = 0.0f;
      if constexpr (HAS_BIAS) badd = bias[col];
      bool to_v = false;
      int vbase = 0;
      if constexpr (SPLIT_V) {
        if (col >= 1536) {
          to_v = true;
          int dg = col - 1536;
          vbase = ((dg >> 6) << 6) * 12 + (dg & 63) +  // placeholder, fixed below
                  0;
          vbase = ((dg >> 6) + 0) * 0;  // (unused; real index computed per-row)
          (void)vbase;
        }
      }
#pragma unroll
      for (int r = 0; r < 8; ++r) {
        int row = m0 + wm + mi * 16 + hi * 8 + r;
        float val = acc[mi][ni][r] + badd;
        if constexpr (SPLIT_V) {
          if (to_v) {
            int dg = col - 1536;
            int hh = dg >> 6, dd = dg & 63;
            int bb = row >> 11, ss = row & 2047;  // S = 2048
            Cv[(size_t)(((bb * 12 + hh) << 6) + dd) * 2048 + ss] = (TC)val;
            continue;
          }
        }
        C[(size_t)row * ldc + col] = (TC)val;
      }
    }
  }
}

// ---------------------------------------------------------------------------
// Flash attention: block = 128 threads (4 waves) = 64 query rows of one (b,h).
// Q/K tiles TDM'd from the f16 qkv workspace; V tiles TDM'd from the
// pre-transposed [b][h][d][s] workspace (lands directly as Vt[d][key]).
// ---------------------------------------------------------------------------
__global__ __launch_bounds__(128) void attn_kernel(
    const _Float16* __restrict__ qkv, const _Float16* __restrict__ vt,
    _Float16* __restrict__ attn_out) {
  constexpr int S = 2048, D3 = 2304, Dh = 64, DIM = 768;
  const float SCALE = 0.125f;  // 64^-0.5

  __shared__ __align__(32) _Float16 Qs[64][64];
  __shared__ __align__(32) _Float16 Ks[32][64];     // K rows (Bt for Q*K^T)
  __shared__ __align__(32) _Float16 Vt[64][32];     // Vt[d][key] (Bt for P*V)
  __shared__ __align__(32) _Float16 Ps[4][16][32];  // per-wave P tile

  const int t    = threadIdx.x;
  const int lane = t & 31;
  const int w    = t >> 5;
  const int hi   = lane >> 4;
  const int ln   = lane & 15;

  int idx = blockIdx.x;
  const int qblk = idx & 31;  // S/64 = 32
  idx >>= 5;
  const int h = idx % 12;
  const int b = idx / 12;

  const size_t rowbase = (size_t)b * S;
  const int qcol = h * Dh;
  const int kcol = DIM + h * Dh;
  const int q0 = qblk * 64;
  const size_t vtbase = (size_t)((b * 12 + h) * Dh) * S;  // vt[b][h][0][0]

  // Stage Q tile (64x64).
#if HAVE_TDM
  if (w == 0) {
    tdm_load_2d_f16(lds_addr_of(&Qs[0][0]),
                    qkv + (rowbase + q0) * D3 + qcol, 64, 64, D3);
    __builtin_amdgcn_s_wait_tensorcnt(0);
  }
#else
#pragma unroll
  for (int i = 0; i < 32; ++i) {
    int linear = i * 128 + t;
    int r = linear >> 6, d = linear & 63;
    Qs[r][d] = qkv[(rowbase + q0 + r) * D3 + qcol + d];
  }
#endif
  __syncthreads();

  v16h qf[2];
  qf[0] = make_a_frag(&Qs[w * 16 + ln][0], hi);   // d = 0..31
  qf[1] = make_a_frag(&Qs[w * 16 + ln][32], hi);  // d = 32..63

  v8f o[4] = {};
  float mrow[8], lrow[8];
#pragma unroll
  for (int r = 0; r < 8; ++r) { mrow[r] = -1e30f; lrow[r] = 0.0f; }

  for (int kb = 0; kb < S; kb += 32) {
    __syncthreads();
#if HAVE_TDM
    if (w == 0) {
      tdm_load_2d_f16(lds_addr_of(&Ks[0][0]),
                      qkv + (rowbase + kb) * D3 + kcol, 64, 32, D3);
      tdm_load_2d_f16(lds_addr_of(&Vt[0][0]), vt + vtbase + kb, 32, 64, S);
      __builtin_amdgcn_s_wait_tensorcnt(0);
    }
#else
#pragma unroll
    for (int i = 0; i < 16; ++i) {
      int linear = i * 128 + t;
      int r = linear >> 6, d = linear & 63;
      Ks[r][d] = qkv[(rowbase + kb + r) * D3 + kcol + d];
      Vt[d][r] = vt[vtbase + (size_t)d * S + kb + r];
    }
#endif
    __syncthreads();

    // S = Q * K^T : 16x32 tile as two 16x16 WMMA outputs, K-dim = 64 (2 steps).
    v8f sacc[2] = {};
#pragma unroll
    for (int ni = 0; ni < 2; ++ni) {
      v16h kf0 = *(const v16h*)&Ks[ni * 16 + ln][hi * 16];       // d 0..31
      v16h kf1 = *(const v16h*)&Ks[ni * 16 + ln][32 + hi * 16];  // d 32..63
      sacc[ni] = WMMA_F32_F16(qf[0], kf0, sacc[ni]);
      sacc[ni] = WMMA_F32_F16(qf[1], kf1, sacc[ni]);
    }

    // Online softmax over the 32 new key columns.
#pragma unroll
    for (int r = 0; r < 8; ++r) {
      float s0 = sacc[0][r] * SCALE;
      float s1 = sacc[1][r] * SCALE;
      float vmax = fmaxf(s0, s1);
      vmax = fmaxf(vmax, __shfl_xor(vmax, 1));
      vmax = fmaxf(vmax, __shfl_xor(vmax, 2));
      vmax = fmaxf(vmax, __shfl_xor(vmax, 4));
      vmax = fmaxf(vmax, __shfl_xor(vmax, 8));
      float newm = fmaxf(mrow[r], vmax);
      float cf = __expf(mrow[r] - newm);
      float p0 = __expf(s0 - newm);
      float p1 = __expf(s1 - newm);
      float psum = p0 + p1;
      psum += __shfl_xor(psum, 1);
      psum += __shfl_xor(psum, 2);
      psum += __shfl_xor(psum, 4);
      psum += __shfl_xor(psum, 8);
      lrow[r] = lrow[r] * cf + psum;
      mrow[r] = newm;
#pragma unroll
      for (int ni = 0; ni < 4; ++ni) o[ni][r] *= cf;
      Ps[w][hi * 8 + r][ln]      = (_Float16)p0;
      Ps[w][hi * 8 + r][16 + ln] = (_Float16)p1;
    }
    __syncthreads();

    // O += P(16x32) * V(32x64): P relaid as an A-fragment, Vt already staged.
    v16h pf = make_a_frag(&Ps[w][ln][0], hi);
#pragma unroll
    for (int ni = 0; ni < 4; ++ni) {
      v16h vf = *(const v16h*)&Vt[ni * 16 + ln][hi * 16];
      o[ni] = WMMA_F32_F16(pf, vf, o[ni]);
    }
  }

  // Normalize and store (f16 intermediate for the projection GEMM).
#pragma unroll
  for (int ni = 0; ni < 4; ++ni) {
#pragma unroll
    for (int r = 0; r < 8; ++r) {
      int row = q0 + w * 16 + hi * 8 + r;
      float val = o[ni][r] / lrow[r];
      attn_out[(rowbase + row) * DIM + h * Dh + ni * 16 + ln] = (_Float16)val;
    }
  }
}

extern "C" void kernel_launch(void* const* d_in, const int* in_sizes, int n_in,
                              void* d_out, int out_size, void* d_ws, size_t ws_size,
                              hipStream_t stream) {
  (void)in_sizes; (void)n_in; (void)out_size; (void)ws_size;
  const float* x     = (const float*)d_in[0];
  const float* Wqkv  = (const float*)d_in[1];
  const float* Wproj = (const float*)d_in[2];
  const float* bproj = (const float*)d_in[3];
  float* out = (float*)d_out;

  const int Brows = 4 * 2048;  // 8192 total token rows
  _Float16* qkv  = (_Float16*)d_ws;                 // 8192*2304 halfs (Q,K live)
  _Float16* vt   = qkv + (size_t)Brows * 2304;      // 8192*768 halfs, [b][h][d][s]
  _Float16* attn = vt + (size_t)Brows * 768;        // 8192*768 halfs

  dim3 blk(256);
  // qkv = x @ Wqkv  (f32 in, f16 out; V third scattered pre-transposed)
  gemm_wmma_kernel<float, float, _Float16, false, true>
      <<<dim3(2304 / 128, Brows / 128), blk, 0, stream>>>(
          x, Wqkv, qkv, vt, nullptr, Brows, 2304, 768, 768, 2304, 2304);

  // attention per (b, h, 64-row q block)
  attn_kernel<<<dim3(4 * 12 * (2048 / 64)), dim3(128), 0, stream>>>(qkv, vt, attn);

  // out = attn @ Wproj + bproj  (f16 A via TDM, f32 B, f32 out)
  gemm_wmma_kernel<_Float16, float, float, true, false>
      <<<dim3(768 / 128, Brows / 128), blk, 0, stream>>>(
          attn, Wproj, out, nullptr, bproj, Brows, 768, 768, 768, 768, 768);
}
